// EuclideanTransformerRelativeAttention_77670188581332
// MI455X (gfx1250) — matrix operand order repaired and checked by reading, not compiled, not run on hardware
//
#include <hip/hip_runtime.h>
#include <hip/hip_bf16.h>

// ---------------- problem constants ----------------
#define BSZ   4
#define S0    1024
#define POOLN 8
#define SREAL 1032
#define SP    1056          // padded seq: 33*32 = 66*16
#define DMD   512
#define NHD   8
#define DHD   64
#define NORMC 13370.9461f   // sqrt(2*1024*333312 - 2*15872^2), closed form
#define GEPS  1e-10f

#define USE_TDM   1
#define USE_ASYNC 1

typedef __attribute__((ext_vector_type(16))) __bf16 v16bf;
typedef __attribute__((ext_vector_type(8)))  float  v8f;
typedef unsigned int v4u __attribute__((ext_vector_type(4)));
typedef unsigned int v8u __attribute__((ext_vector_type(8)));

union A16u {
    v16bf v;
    unsigned short u[16];
    uint4 q[2];
};

__device__ __forceinline__ float bf2f(unsigned short u) {
    union { unsigned int i; float f; } x; x.i = ((unsigned int)u) << 16; return x.f;
}
__device__ __forceinline__ unsigned short f2bf(float f) {
    union { float f; unsigned int i; } x; x.f = f;
    unsigned int i = x.i;
    unsigned int r = (i + 0x7FFFu + ((i >> 16) & 1u)) >> 16;   // RNE
    return (unsigned short)r;
}

// A-matrix 16x32 bf16 tile: lane = M row (lane&15); two contiguous 16B chunks.
__device__ __forceinline__ v16bf loadA16(const unsigned short* rowk, int lane) {
    int off0 = (lane >> 4) * 8;
    A16u t;
    t.q[0] = *(const uint4*)(rowk + off0);
    t.q[1] = *(const uint4*)(rowk + off0 + 16);
    return t.v;
}
// B-matrix 32x16 bf16 tile: lane = N col; two contiguous 16B chunks (same K order as A).
__device__ __forceinline__ v16bf loadB16(const unsigned short* colk, int lane) {
    int off = (lane >> 4) * 8;
    A16u t;
    t.q[0] = *(const uint4*)(colk + off);
    t.q[1] = *(const uint4*)(colk + off + 16);
    return t.v;
}

__device__ __forceinline__ v8f wmma_bf16(v16bf a, v16bf b, v8f c) {
    return __builtin_amdgcn_wmma_f32_16x16x32_bf16(false, a, false, b, (short)0, c, false, false);
}

// TDM: DMA a [16 rows x 512 bf16] row-major tile (stride 512 elems) into LDS.
// D# per CDNA5 ISA ch.8: group0 {count,lds_addr,global_addr,type=2},
// group1 {data_size=2B, tensor_dim0=512, tensor_dim1=16, tile 512x16, stride0=512}.
__device__ __forceinline__ void tdm_load_rows16(const unsigned short* gsrc, unsigned lds_byte_addr) {
#if USE_TDM
    unsigned long long ga = (unsigned long long)(size_t)gsrc;
    v4u g0 = { 1u,                                    // count=1, is_restore=0
               lds_byte_addr,
               (unsigned)ga,
               ((unsigned)(ga >> 32) & 0x1FFFFFFu) | (2u << 30) };  // type=2 (image)
    v8u g1 = { (1u << 16),        // workgroup_mask=0, data_size=1 (2 bytes)
               (512u << 16),      // tensor_dim0[15:0]=512
               (16u << 16),       // tensor_dim0 hi=0, tensor_dim1[15:0]=16
               (512u << 16),      // tensor_dim1 hi=0, tile_dim0=512
               16u,               // tile_dim1=16, tile_dim2=0
               512u,              // tensor_dim0_stride lo32 = 512
               0u, 0u };
    asm volatile("tensor_load_to_lds %0, %1" :: "s"(g0), "s"(g1) : "memory");
    __builtin_amdgcn_s_wait_tensorcnt(0);
#endif
}

// directional weight: k=0 relu(+dh), 1 relu(+dw), 2 relu(-dh), 3 relu(-dw); 0.25 outside s0 block
__device__ __forceinline__ float vdir(int dir, int qi, int j) {
    if (qi < S0 && j < S0) {
        float d = (dir & 1) ? (float)((qi & 31) - (j & 31))
                            : (float)((qi >> 5) - (j >> 5));
        float r = (d + GEPS) / (NORMC + GEPS);
        if (dir >= 2) r = -r;
        return fmaxf(r, 0.0f);
    }
    return 0.25f;
}

// ============================================================
// K0a: pack x = concat(h, h_pool) -> bf16 [B][SP][DM] (zero pad)
// ============================================================
__global__ void pack_x_kernel(const float* __restrict__ h, const float* __restrict__ pool,
                              unsigned short* __restrict__ xb) {
    size_t idx = (size_t)blockIdx.x * 256 + threadIdx.x;
    size_t total = (size_t)BSZ * SP * DMD;
    if (idx >= total) return;
    int col = (int)(idx & 511u);
    size_t t = idx >> 9;
    int sp = (int)(t % SP);
    int b  = (int)(t / SP);
    float v = 0.0f;
    if (sp < S0)            v = h[((size_t)b * S0 + sp) * DMD + col];
    else if (sp < SREAL)    v = pool[((size_t)b * POOLN + (sp - S0)) * DMD + col];
    xb[idx] = f2bf(v);
}

// ============================================================
// K0b: weights -> wT [3584][512] bf16:
//   rows 0..511 qT[nd][h], 512..1023 kT, 1024..3071 vT[(dir,nd)][h], 3072..3583 o[h][nd]
// ============================================================
__global__ void pack_w_kernel(const float* __restrict__ q, const float* __restrict__ k,
                              const float* __restrict__ v, const float* __restrict__ o,
                              unsigned short* __restrict__ wT) {
    size_t idx = (size_t)blockIdx.x * 256 + threadIdx.x;
    size_t total = (size_t)3584 * 512;
    if (idx >= total) return;
    int x = (int)(idx & 511u);
    int c = (int)(idx >> 9);
    float val;
    if (c < 512)        val = q[(size_t)x * 512 + c];
    else if (c < 1024)  val = k[(size_t)x * 512 + (c - 512)];
    else if (c < 3072)  val = v[(size_t)x * 2048 + (c - 1024)];
    else                val = o[(size_t)(c - 3072) * 512 + x];
    wT[idx] = f2bf(val);
}

// ============================================================
// K1: projections. x row-tile staged to LDS by TDM once per WGP;
//   192 col-tiles of 16 over {Q(512) K(512) V(2048)}.
// ============================================================
__global__ void proj_kernel(const unsigned short* __restrict__ xb,
                            const unsigned short* __restrict__ wT,
                            unsigned short* __restrict__ Qb,
                            unsigned short* __restrict__ Kb,
                            unsigned short* __restrict__ Vt) {
    extern __shared__ unsigned char smem[];
    unsigned short* sX = (unsigned short*)smem;          // [16][512] bf16

    int blk = blockIdx.x;
    int b  = blk / (SP / 16);
    int rt = blk % (SP / 16);
    int w = threadIdx.x >> 5, lane = threadIdx.x & 31;
    int r0 = rt * 16;

    const unsigned short* gsrc = xb + ((size_t)b * SP + r0) * DMD;
#if USE_TDM
    if (w == 0) tdm_load_rows16(gsrc, (unsigned)(size_t)sX);
#else
    for (int t = threadIdx.x; t < 16 * DMD / 8; t += 512)
        ((uint4*)sX)[t] = ((const uint4*)gsrc)[t];
#endif
    __syncthreads();

    const unsigned short* arow = sX + (size_t)(lane & 15) * DMD;

    for (int ct = w; ct < 192; ct += 16) {
        v8f c = {};
        int col0 = ct * 16;
        const unsigned short* brow = wT + (size_t)(col0 + (lane & 15)) * DMD;
        #pragma unroll 4
        for (int kb = 0; kb < DMD; kb += 32) {
            v16bf a  = loadA16(arow + kb, lane);
            v16bf bm = loadB16(brow + kb, lane);
            c = wmma_bf16(a, bm, c);
        }
        int m8 = 8 * (lane >> 4);
        int ccol = col0 + (lane & 15);
        // col tiles never straddle the 512/1024 boundaries -> one branch per tile
        if (col0 < 512) {
            for (int vv = 0; vv < 8; vv++)
                Qb[((size_t)b * SP + r0 + m8 + vv) * 512 + ccol] = f2bf(c[vv]);
        } else if (col0 < 1024) {
            for (int vv = 0; vv < 8; vv++)
                Kb[((size_t)b * SP + r0 + m8 + vv) * 512 + (ccol - 512)] = f2bf(c[vv]);
        } else {
            int cc = ccol - 1024;
            int dir = cc >> 9, nd = cc & 511;
            unsigned short* vbase = Vt + (((size_t)b * 4 + dir) * 512 + nd) * SP + r0 + m8;
            for (int vv = 0; vv < 8; vv++) vbase[vv] = f2bf(c[vv]);
        }
    }
}

// ============================================================
// K2: attention for one (b, head, 16-query tile).
//  Pass1: WMMA scores + analytic bias/mask -> LDS fp32 [16][SP]
//  Pass2: exact softmax per row (wave per row) -> LDS bf16 P
//  Pass3: 4-direction reweighted P@V, V B-tiles software-pipelined
//         through LDS with global_load_async_to_lds_b128 (per-wave ASYNCcnt)
// ============================================================
#define ATTN_SC_OFF  0
#define ATTN_P_OFF   (16 * SP * 4)
#define ATTN_STG_OFF (16 * SP * 4 + 16 * SP * 2)
#define ATTN_LDS     (ATTN_STG_OFF + 4 * 8192)

__global__ void attn_kernel(const unsigned short* __restrict__ Qb,
                            const unsigned short* __restrict__ Kb,
                            const unsigned short* __restrict__ Vt,
                            unsigned short* __restrict__ Red) {
    extern __shared__ unsigned char smem[];
    float*          sS = (float*)(smem + ATTN_SC_OFF);          // 16*SP fp32
    unsigned short* sP = (unsigned short*)(smem + ATTN_P_OFF);  // 16*SP bf16

    int blk = blockIdx.x;
    int b   = blk / (NHD * 65);
    int rem = blk % (NHD * 65);
    int n   = rem / 65;
    int qt  = rem % 65;
    int w = threadIdx.x >> 5, lane = threadIdx.x & 31;
    float slope = exp2f(-(float)(n + 1));

    // ---- pass 1: scores ----
    const unsigned short* qrow =
        Qb + ((size_t)b * SP + qt * 16 + (lane & 15)) * 512 + n * DHD;
    for (int kt = w; kt < SP / 16; kt += 16) {
        v8f c = {};
        const unsigned short* krow =
            Kb + ((size_t)b * SP + kt * 16 + (lane & 15)) * 512 + n * DHD;
        #pragma unroll
        for (int kb = 0; kb < DHD; kb += 32) {
            v16bf a  = loadA16(qrow + kb, lane);
            v16bf bm = loadB16(krow + kb, lane);
            c = wmma_bf16(a, bm, c);
        }
        int m8 = 8 * (lane >> 4);
        int j  = kt * 16 + (lane & 15);
        for (int vv = 0; vv < 8; vv++) {
            int qi = qt * 16 + m8 + vv;
            bool allowed;
            if (qi < S0)            allowed = (j < S0);
            else if (qi < SREAL)    allowed = (j < S0) || (j == qi);
            else                    allowed = (j == 0);   // keep padded rows finite
            float val;
            if (allowed) {
                float bias = 0.0f;
                if (qi < S0 && j < S0) {
                    float fh = (float)((qi >> 5) - (j >> 5));
                    float fw = (float)((qi & 31) - (j & 31));
                    bias = sqrtf(fh * fh + fw * fw) * slope;
                }
                val = c[vv] - bias;
            } else {
                val = -3.0e38f;
            }
            sS[(m8 + vv) * SP + j] = val;
        }
    }
    __syncthreads();

    // ---- pass 2: softmax (wave w owns row w) ----
    {
        int r = w;
        float mx = -3.0e38f;
        for (int t = lane; t < SP; t += 32) mx = fmaxf(mx, sS[r * SP + t]);
        for (int o2 = 16; o2 > 0; o2 >>= 1) mx = fmaxf(mx, __shfl_xor(mx, o2));
        float sum = 0.0f;
        for (int t = lane; t < SP; t += 32) sum += __expf(sS[r * SP + t] - mx);
        for (int o2 = 16; o2 > 0; o2 >>= 1) sum += __shfl_xor(sum, o2);
        float inv = 1.0f / sum;
        for (int t = lane; t < SP; t += 32)
            sP[r * SP + t] = f2bf(__expf(sS[r * SP + t] - mx) * inv);
    }
    __syncthreads();

    // ---- pass 3: waves 0..3 each own one 16-wide d block ----
    if (w < 4) {
        int dblk = w;
        int m = lane & 15;
        int qi = qt * 16 + m;
        unsigned ldsbase = (unsigned)(size_t)smem;   // LDS byte address of dyn-smem base
        unsigned stg0 = (unsigned)(ATTN_STG_OFF + w * 8192);

#if USE_ASYNC
        // async-stage the 4 direction B-tiles ([16 d x 32 j] bf16 each) for key-tile kt
        auto issue_tiles = [&](int kt, int buf) {
            #pragma unroll
            for (int dir = 0; dir < 4; dir++) {
                const unsigned short* gsrc =
                    Vt + (((size_t)b * 4 + dir) * 512 + n * DHD + dblk * 16) * SP + kt * 32;
                unsigned long long gbase = (unsigned long long)(size_t)gsrc;
                int c0 = lane, c1 = lane + 32;          // 64 16B chunks per dir tile
                unsigned voff0 = (unsigned)((c0 >> 2) * (SP * 2) + (c0 & 3) * 16);
                unsigned voff1 = (unsigned)((c1 >> 2) * (SP * 2) + (c1 & 3) * 16);
                unsigned l0 = ldsbase + stg0 + buf * 4096 + dir * 1024
                            + (unsigned)((c0 >> 2) * 64 + (c0 & 3) * 16);
                unsigned l1 = ldsbase + stg0 + buf * 4096 + dir * 1024
                            + (unsigned)((c1 >> 2) * 64 + (c1 & 3) * 16);
                asm volatile("global_load_async_to_lds_b128 %0, %1, %2"
                             :: "v"(l0), "v"(voff0), "s"(gbase) : "memory");
                asm volatile("global_load_async_to_lds_b128 %0, %1, %2"
                             :: "v"(l1), "v"(voff1), "s"(gbase) : "memory");
            }
        };
        issue_tiles(0, 0);
#endif
        v8f acc = {};
        for (int kt = 0; kt < SP / 32; kt++) {
#if USE_ASYNC
            if (kt + 1 < SP / 32) {
                issue_tiles(kt + 1, (kt + 1) & 1);
                asm volatile("s_wait_asynccnt 0x8" ::: "memory");   // oldest 8 (this kt) done
            } else {
                asm volatile("s_wait_asynccnt 0x0" ::: "memory");
            }
            const unsigned short* stg =
                (const unsigned short*)(smem + ATTN_STG_OFF + w * 8192 + (kt & 1) * 4096);
#endif
            int jbase = kt * 32 + (lane >> 4) * 8;
            float p[16];
            int   jj[16];
            #pragma unroll
            for (int e = 0; e < 16; e++) {
                int je = jbase + (e < 8 ? e : e + 8);
                jj[e] = je;
                p[e]  = bf2f(sP[m * SP + je]);
            }
            #pragma unroll
            for (int dir = 0; dir < 4; dir++) {
                A16u au;
                #pragma unroll
                for (int e = 0; e < 16; e++)
                    au.u[e] = f2bf(p[e] * vdir(dir, qi, jj[e]));
#if USE_ASYNC
                // staged layout: [dir][16 d-rows][32 j] bf16 -> B tile via ds_load_b128
                v16bf bm = loadB16(stg + dir * 512 + (lane & 15) * 32, lane);
#else
                const unsigned short* vrow =
                    Vt + (((size_t)b * 4 + dir) * 512 + n * DHD + dblk * 16 + (lane & 15)) * SP
                       + kt * 32;
                v16bf bm = loadB16(vrow, lane);
#endif
                acc = wmma_bf16(au.v, bm, acc);
            }
        }
        int m8 = 8 * (lane >> 4);
        int col = n * DHD + dblk * 16 + (lane & 15);
        for (int vv = 0; vv < 8; vv++) {
            int row = qt * 16 + m8 + vv;
            Red[((size_t)b * SP + row) * 512 + col] = f2bf(acc[vv]);
        }
    }
}

// ============================================================
// K3: out = reduced @ o (+ residual x) -> LayerNorm -> fp32.
//  Red row-tile staged to LDS via TDM (reused by 16 waves x 2 col-tiles).
// ============================================================
#define OUT_Y_OFF  0
#define OUT_R_OFF  (16 * 512 * 4)
#define OUT_LDS    (OUT_R_OFF + 16 * 512 * 2)

__global__ void out_kernel(const unsigned short* __restrict__ Red,
                           const unsigned short* __restrict__ wT,   // rows 3072.. = o[h][nd]
                           const float* __restrict__ h, const float* __restrict__ pool,
                           const float* __restrict__ gamma, const float* __restrict__ beta,
                           float* __restrict__ out) {
    extern __shared__ unsigned char smem[];
    float*          sY = (float*)(smem + OUT_Y_OFF);            // [16][512] fp32
    unsigned short* sR = (unsigned short*)(smem + OUT_R_OFF);   // [16][512] bf16

    int blk = blockIdx.x;
    int b  = blk / 65;
    int rt = blk % 65;
    int r0 = rt * 16;
    int w = threadIdx.x >> 5, lane = threadIdx.x & 31;

    const unsigned short* gsrc = Red + ((size_t)b * SP + r0) * 512;
#if USE_TDM
    if (w == 0) tdm_load_rows16(gsrc, (unsigned)(size_t)sR);
#else
    for (int t = threadIdx.x; t < 16 * 512 / 8; t += 512)
        ((uint4*)sR)[t] = ((const uint4*)gsrc)[t];
#endif
    __syncthreads();

    const unsigned short* arow = sR + (size_t)(lane & 15) * 512;
    for (int tctr = 0; tctr < 2; tctr++) {
        int ct = 2 * w + tctr;          // 32 col tiles over DM=512
        v8f c = {};
        const unsigned short* brow = wT + (size_t)(3072 + ct * 16 + (lane & 15)) * 512;
        #pragma unroll 4
        for (int kb = 0; kb < 512; kb += 32) {
            v16bf a  = loadA16(arow + kb, lane);
            v16bf bm = loadB16(brow + kb, lane);
            c = wmma_bf16(a, bm, c);
        }
        int m8 = 8 * (lane >> 4);
        int col = ct * 16 + (lane & 15);
        for (int vv = 0; vv < 8; vv++) sY[(m8 + vv) * 512 + col] = c[vv];
    }
    __syncthreads();

    // LayerNorm: wave w owns row w
    int r = w;
    int gi = r0 + r;
    if (gi < SREAL) {
        float xs[16];
        float s = 0.0f, s2 = 0.0f;
        #pragma unroll
        for (int i = 0; i < 16; i++) {
            int col = lane + 32 * i;
            float xv = (gi < S0) ? h[((size_t)b * S0 + gi) * DMD + col]
                                 : pool[((size_t)b * POOLN + (gi - S0)) * DMD + col];
            float y = sY[r * 512 + col] + xv;
            xs[i] = y; s += y; s2 += y * y;
        }
        for (int o2 = 16; o2 > 0; o2 >>= 1) { s += __shfl_xor(s, o2); s2 += __shfl_xor(s2, o2); }
        float mu  = s * (1.0f / 512.0f);
        float var = s2 * (1.0f / 512.0f) - mu * mu;
        float rstd = rsqrtf(var + 1e-12f);
        #pragma unroll
        for (int i = 0; i < 16; i++) {
            int col = lane + 32 * i;
            out[((size_t)b * SREAL + gi) * DMD + col] =
                (xs[i] - mu) * rstd * gamma[col] + beta[col];
        }
    }
}

// ============================================================
extern "C" void kernel_launch(void* const* d_in, const int* in_sizes, int n_in,
                              void* d_out, int out_size, void* d_ws, size_t ws_size,
                              hipStream_t stream) {
    const float* h     = (const float*)d_in[0];
    const float* hpool = (const float*)d_in[1];
    const float* q     = (const float*)d_in[2];
    const float* k     = (const float*)d_in[3];
    const float* v     = (const float*)d_in[4];
    const float* o     = (const float*)d_in[5];
    const float* gamma = (const float*)d_in[6];
    const float* beta  = (const float*)d_in[7];
    float* out = (float*)d_out;

    // workspace layout (ushort elements)
    const size_t N_XB  = (size_t)BSZ * SP * DMD;       // 2,162,688
    const size_t N_W   = (size_t)3584 * 512;           // 1,835,008
    const size_t N_VT  = (size_t)BSZ * 4 * 512 * SP;   // 8,650,752
    unsigned short* ws = (unsigned short*)d_ws;
    unsigned short* xb  = ws;
    unsigned short* wT  = xb + N_XB;
    unsigned short* Qb  = wT + N_W;
    unsigned short* Kb  = Qb + N_XB;
    unsigned short* Vt  = Kb + N_XB;
    unsigned short* Red = Vt + N_VT;

    (void)hipFuncSetAttribute((const void*)attn_kernel,
                              hipFuncAttributeMaxDynamicSharedMemorySize, ATTN_LDS);
    (void)hipFuncSetAttribute((const void*)out_kernel,
                              hipFuncAttributeMaxDynamicSharedMemorySize, OUT_LDS);

    {
        size_t total = (size_t)BSZ * SP * DMD;
        pack_x_kernel<<<(unsigned)((total + 255) / 256), 256, 0, stream>>>(h, hpool, xb);
    }
    {
        size_t total = (size_t)3584 * 512;
        pack_w_kernel<<<(unsigned)((total + 255) / 256), 256, 0, stream>>>(q, k, v, o, wT);
    }
    proj_kernel<<<BSZ * (SP / 16), 512, 16 * DMD * 2, stream>>>(xb, wT, Qb, Kb, Vt);
    attn_kernel<<<BSZ * NHD * 65, 512, ATTN_LDS, stream>>>(Qb, Kb, Vt, Red);
    out_kernel<<<BSZ * 65, 512, OUT_LDS, stream>>>(Red, wT, h, hpool, gamma, beta, out);
}
